// DNBDeep_41317585388073
// MI455X (gfx1250) — matrix operand order repaired and checked by reading, not compiled, not run on hardware
//
#include <hip/hip_runtime.h>

// ---------------------------------------------------------------------------
// MI455X (gfx1250) GAT pipeline.
//  - Dense GEMMs: wave32 WMMA bf16 (v_wmma_f32_16x16x32_bf16), f32 accumulate.
//    Fragments loaded with unguarded 128-bit vector loads (A: row-clamped +
//    K zero-padded operands; B: pre-transposed bf16 weights Wt[Nc][Kpad]).
//  - Edge softmax/scatter: streaming atomics (bandwidth bound on 23.3 TB/s).
// ---------------------------------------------------------------------------

typedef __bf16 bf16_t;
typedef __attribute__((ext_vector_type(16))) __bf16 v16bf;
typedef __attribute__((ext_vector_type(8)))  float  v8f;

#define NEG_ATTN 0.2f
#define NEG_ACT  0.01f

// monotonic float <-> uint encoding so atomicMax(u32) == float max
__device__ __forceinline__ unsigned fenc(float f) {
  unsigned u = __float_as_uint(f);
  return (u & 0x80000000u) ? ~u : (u | 0x80000000u);
}
__device__ __forceinline__ float fdec(unsigned u) {
  return __uint_as_float((u & 0x80000000u) ? (u & 0x7fffffffu) : ~u);
}

// ------------------------------- fills -------------------------------------
__global__ void k_fill_f32(float* p, float v, long n) {
  long i = (long)blockIdx.x * blockDim.x + threadIdx.x;
  if (i < n) p[i] = v;
}
__global__ void k_fill_u32(unsigned* p, unsigned v, long n) {
  long i = (long)blockIdx.x * blockDim.x + threadIdx.x;
  if (i < n) p[i] = v;
}

// W [K x Nc] f32 (row-major)  ->  Wt [Nc x Kpad] bf16, zero-padded K tail.
__global__ void k_prep_w(const float* __restrict__ W, bf16_t* __restrict__ Wt,
                         int K, int Nc, int Kpad) {
  long t = (long)blockIdx.x * blockDim.x + threadIdx.x;
  if (t >= (long)Nc * Kpad) return;
  const int c = (int)(t / Kpad), k = (int)(t % Kpad);
  Wt[t] = (k < K) ? (bf16_t)W[(size_t)k * Nc + c] : (bf16_t)0.0f;
}

// ef [E x 16] -> efp [E x 32] with zeroed pad columns
__global__ void k_pad_copy(const float* __restrict__ ef, float* __restrict__ efp, int E) {
  long t = (long)blockIdx.x * blockDim.x + threadIdx.x;
  if (t >= (long)E * 32) return;
  const int e = (int)(t >> 5), f = (int)(t & 31);
  efp[t] = (f < 16) ? ef[(size_t)e * 16 + f] : 0.0f;
}

// ---------------------------------------------------------------------------
// WMMA GEMM: OUT[M x Nc] (stride ldc) = act(X[M x K](stride lda) @ W + bias)
// W supplied pre-transposed as Wt[Nc x Kpad] bf16. Requirements:
//   Kpad % 32 == 0, lda >= Kpad with X's K-tail zero-filled, Nc % 16 == 0,
//   X and Wt 32-byte aligned, lda % 8 == 0.
// One wave per 16x16 tile. Per k-step: A = two v8f loads (2x b128 each),
// B = one v16bf load (2x b128), 8x cvt_pk_bf16, 1x wmma. No guards before
// the WMMA (row/col clamped), so EXEC is all-ones.
// ---------------------------------------------------------------------------
__global__ void k_gemm(const float* __restrict__ X, int lda,
                       const bf16_t* __restrict__ Wt, int Kpad,
                       const float* __restrict__ bias, float* __restrict__ OUT,
                       int M, int Nc, int ldc, int act) {
  const int lane = threadIdx.x & 31;
  const int half = lane >> 4;
  const int lm   = lane & 15;
  const int row  = blockIdx.x * 16 + lm;
  const int col  = blockIdx.y * 16 + lm;
  const int srow = (row < M) ? row : (M - 1);      // clamped: loads always valid
  const int scol = (col < Nc) ? col : (Nc - 1);
  const float*  Xr = X  + (size_t)srow * lda;
  const bf16_t* Wr = Wt + (size_t)scol * Kpad;

  v8f acc = {};
  for (int k0 = 0; k0 < Kpad; k0 += 32) {
    const v8f r0 = *(const v8f*)(Xr + k0 + half * 16);       // K = k0+half*16 .. +7
    const v8f r1 = *(const v8f*)(Xr + k0 + half * 16 + 8);   // K = k0+half*16+8 .. +15
    const v16bf b = *(const v16bf*)(Wr + k0 + half * 16);    // K = k0+half*16 .. +15
    v16bf a;
#pragma unroll
    for (int j = 0; j < 8; ++j) a[j] = (bf16_t)r0[j];
#pragma unroll
    for (int j = 0; j < 8; ++j) a[8 + j] = (bf16_t)r1[j];
    acc = __builtin_amdgcn_wmma_f32_16x16x32_bf16(
        /*neg_a=*/false, a, /*neg_b=*/false, b,
        /*c_mod=*/(short)0, acc, /*reuse_a=*/false, /*reuse_b=*/false);
  }

  const float bb = (bias != nullptr) ? bias[scol] : 0.0f;
  const bool colOK = col < Nc;                      // uniform per wave (Nc % 16 == 0)
#pragma unroll
  for (int r = 0; r < 8; ++r) {
    const int orow = blockIdx.x * 16 + r + half * 8;
    if (orow < M && colOK) {
      float v = acc[r] + bb;
      if (act == 1)      v = fmaxf(v, 0.0f);
      else if (act == 2) v = (v > 0.0f) ? v : NEG_ACT * v;
      OUT[(size_t)orow * ldc + col] = v;
    }
  }
}

// ----------------------- edge / scatter kernels ----------------------------
// concat(hn[src], he) scatter-summed over dst -> xsum[N, 96] (cols 80..95 pad)
__global__ void k_concat_scatter(const int* __restrict__ src, const int* __restrict__ dst,
                                 const float* __restrict__ hn, const float* __restrict__ he,
                                 float* __restrict__ xsum, int E) {
  long t = (long)blockIdx.x * blockDim.x + threadIdx.x;
  if (t >= (long)E * 80) return;
  const int e = (int)(t / 80), f = (int)(t % 80);
  const float v = (f < 64) ? hn[(size_t)src[e] * 64 + f] : he[(size_t)e * 16 + (f - 64)];
  atomicAdd(&xsum[(size_t)dst[e] * 96 + f], v);
}

// el[n,h] = sum_d z[n,h,d]*al[h,d]; er likewise (H=4, D=128)
__global__ void k_attn_logits(const float* __restrict__ z, const float* __restrict__ al,
                              const float* __restrict__ ar, float* __restrict__ el,
                              float* __restrict__ er, int Nn) {
  const int t = blockIdx.x * blockDim.x + threadIdx.x;
  if (t >= Nn * 4) return;
  const int n = t >> 2, h = t & 3;
  const float* zp = z + (size_t)n * 512 + h * 128;
  const float* ap = al + h * 128;
  const float* bp = ar + h * 128;
  float sl = 0.f, sr = 0.f;
  for (int d = 0; d < 128; ++d) { const float zv = zp[d]; sl += zv * ap[d]; sr += zv * bp[d]; }
  el[t] = sl; er[t] = sr;
}

// ev = leaky(el[src]+er[dst], 0.2); segment-max over dst via encoded atomicMax
__global__ void k_edge_logit_max(const int* __restrict__ src, const int* __restrict__ dst,
                                 const float* __restrict__ el, const float* __restrict__ er,
                                 float* __restrict__ ev, unsigned* __restrict__ mx, int E) {
  long t = (long)blockIdx.x * blockDim.x + threadIdx.x;
  if (t >= (long)E * 4) return;
  const int e = (int)(t >> 2), h = (int)(t & 3);
  float x = el[(size_t)src[e] * 4 + h] + er[(size_t)dst[e] * 4 + h];
  x = (x > 0.f) ? x : NEG_ATTN * x;
  ev[t] = x;
  atomicMax(&mx[(size_t)dst[e] * 4 + h], fenc(x));
}

// ev = exp(ev - max[dst]); den[dst] += ev
__global__ void k_edge_exp_sum(const int* __restrict__ dst, float* __restrict__ ev,
                               const unsigned* __restrict__ mx, float* __restrict__ den, int E) {
  long t = (long)blockIdx.x * blockDim.x + threadIdx.x;
  if (t >= (long)E * 4) return;
  const int e = (int)(t >> 2), h = (int)(t & 3);
  const float ex = __expf(ev[t] - fdec(mx[(size_t)dst[e] * 4 + h]));
  ev[t] = ex;
  atomicAdd(&den[(size_t)dst[e] * 4 + h], ex);
}

// ev = ev / den[dst]  (finalize attention coefficients)
__global__ void k_edge_norm(const int* __restrict__ dst, float* __restrict__ ev,
                            const float* __restrict__ den, int E) {
  long t = (long)blockIdx.x * blockDim.x + threadIdx.x;
  if (t >= (long)E * 4) return;
  const int e = (int)(t >> 2), h = (int)(t & 3);
  ev[t] = ev[t] / den[(size_t)dst[e] * 4 + h];
}

// rst[dst, h, d] += ev[e,h] * z[src, h, d]   (dominant bandwidth kernel)
__global__ void k_edge_scatter(const int* __restrict__ src, const int* __restrict__ dst,
                               const float* __restrict__ ev,
                               const float* __restrict__ z, float* __restrict__ rst, int E) {
  long t = (long)blockIdx.x * blockDim.x + threadIdx.x;
  if (t >= (long)E * 512) return;
  const int e = (int)(t >> 9), f = (int)(t & 511), h = f >> 7;
  const int s = src[e], d = dst[e];
  atomicAdd(&rst[(size_t)d * 512 + f], ev[(size_t)e * 4 + h] * z[(size_t)s * 512 + f]);
}

// rst = leaky(rst + gb[f], 0.01)
__global__ void k_bias_leaky(float* __restrict__ rst, const float* __restrict__ gb, long n) {
  long t = (long)blockIdx.x * blockDim.x + threadIdx.x;
  if (t >= n) return;
  const int f = (int)(t & 511);
  const float v = rst[t] + gb[f];
  rst[t] = (v > 0.f) ? v : NEG_ACT * v;
}

// gate[n] = sigmoid(x[n,:] . ws_w + ws_b)
__global__ void k_gate(const float* __restrict__ x, const float* __restrict__ wsw,
                       const float* __restrict__ wsb, float* __restrict__ gate, int Nn) {
  const int n = blockIdx.x * blockDim.x + threadIdx.x;
  if (n >= Nn) return;
  const float* xp = x + (size_t)n * 128;
  float s = 0.f;
  for (int d = 0; d < 128; ++d) s += xp[d] * wsw[d];
  s += wsb[0];
  gate[n] = 1.0f / (1.0f + __expf(-s));
}

// weighted segment-sum + segment-max over graph ids
__global__ void k_readout(const int* __restrict__ gid, const float* __restrict__ x,
                          const float* __restrict__ gate, float* __restrict__ hs,
                          unsigned* __restrict__ hm, int Nn) {
  long t = (long)blockIdx.x * blockDim.x + threadIdx.x;
  if (t >= (long)Nn * 128) return;
  const int n = (int)(t >> 7), d = (int)(t & 127);
  const float xv = x[(size_t)n * 128 + d];
  const int g = gid[n];
  atomicAdd(&hs[(size_t)g * 128 + d], xv * gate[n]);
  atomicMax(&hm[(size_t)g * 128 + d], fenc(xv));
}

__global__ void k_build_hg(const float* __restrict__ hs, const unsigned* __restrict__ hm,
                           float* __restrict__ hg, int B) {
  const int t = blockIdx.x * blockDim.x + threadIdx.x;
  if (t >= B * 256) return;
  const int b = t >> 8, j = t & 255;
  hg[t] = (j < 128) ? hs[b * 128 + j] : fdec(hm[b * 128 + (j - 128)]);
}

// out[b] = tb[b,:] . Wo2 + bo2
__global__ void k_final(const float* __restrict__ tb, const float* __restrict__ Wo2,
                        const float* __restrict__ bo2, float* __restrict__ out, int B) {
  const int b = blockIdx.x * blockDim.x + threadIdx.x;
  if (b >= B) return;
  const float* tp = tb + (size_t)b * 128;
  float s = 0.f;
  for (int d = 0; d < 128; ++d) s += tp[d] * Wo2[d];
  out[b] = s + bo2[0];
}

// ---------------------------------------------------------------------------
extern "C" void kernel_launch(void* const* d_in, const int* in_sizes, int n_in,
                              void* d_out, int out_size, void* d_ws, size_t ws_size,
                              hipStream_t stream) {
  (void)n_in; (void)ws_size;
  const int E  = in_sizes[0];   // 400000
  const int Nn = in_sizes[2];   // 25000
  const int B  = out_size;      // 64

  const int*   srcA = (const int*)d_in[0];
  const int*   dstA = (const int*)d_in[1];
  const int*   gidA = (const int*)d_in[2];
  const float* nfA  = (const float*)d_in[3];
  const float* efA  = (const float*)d_in[4];
  const int*   srcB = (const int*)d_in[5];
  const int*   dstB = (const int*)d_in[6];
  const int*   gidB = (const int*)d_in[7];
  const float* nfB  = (const float*)d_in[8];
  const float* efB  = (const float*)d_in[9];
  const float* p_Wn  = (const float*)d_in[10];
  const float* p_bn  = (const float*)d_in[11];
  const float* p_We  = (const float*)d_in[12];
  const float* p_be  = (const float*)d_in[13];
  const float* p_Wc  = (const float*)d_in[14];
  const float* p_bc  = (const float*)d_in[15];
  const float* p_fc1 = (const float*)d_in[16];
  const float* p_al1 = (const float*)d_in[17];
  const float* p_ar1 = (const float*)d_in[18];
  const float* p_gb1 = (const float*)d_in[19];
  const float* p_Wl1 = (const float*)d_in[20];
  const float* p_bl1 = (const float*)d_in[21];
  const float* p_fc2 = (const float*)d_in[22];
  const float* p_al2 = (const float*)d_in[23];
  const float* p_ar2 = (const float*)d_in[24];
  const float* p_gb2 = (const float*)d_in[25];
  const float* p_Wl2 = (const float*)d_in[26];
  const float* p_bl2 = (const float*)d_in[27];
  const float* p_wsw = (const float*)d_in[28];
  const float* p_wsb = (const float*)d_in[29];
  const float* p_Wp  = (const float*)d_in[30];
  const float* p_bp  = (const float*)d_in[31];
  const float* Wo1   = (const float*)d_in[32];
  const float* bo1   = (const float*)d_in[33];
  const float* Wo2   = (const float*)d_in[34];
  const float* bo2   = (const float*)d_in[35];

  // workspace layout (fp32 words), 32-byte aligned carves, aliased regions
  float* ws = (float*)d_ws;
  size_t o = 0;
  auto take = [&](size_t n) { float* p = ws + o; o += (n + 7) & ~(size_t)7; return p; };
  // region R1: efp [E x 32]  (dead after he gemm)  ALIASED with z [Nn x 512]
  float* R1 = take((size_t)E * 32 > (size_t)Nn * 512 ? (size_t)E * 32 : (size_t)Nn * 512);
  float* efp = R1;
  float* z   = R1;
  // region R2: he [E x 16]  (dead after concat_scatter) ALIASED with attn scratch
  float* R2 = take((size_t)E * 16);
  float* he  = R2;
  float*    ev   = R2;                         // E*4
  float*    el   = R2 + (size_t)E * 4;         // Nn*4
  float*    er   = el + (size_t)Nn * 4;
  unsigned* mx   = (unsigned*)(er + (size_t)Nn * 4);
  float*    den  = er + (size_t)Nn * 8;
  float*    gate = den + (size_t)Nn * 4;       // Nn
  float*    hn   = take((size_t)Nn * 64);
  float*    xsum = take((size_t)Nn * 96);      // stride 96, cols 80..95 zero pad
  float*    xc   = take((size_t)Nn * 96);
  float*    rst  = take((size_t)Nn * 512);
  float*    x1   = take((size_t)Nn * 128);
  float*    x2   = take((size_t)Nn * 128);
  bf16_t*   Wt   = (bf16_t*)take(65536);       // transposed bf16 weights (128 KB)
  float*    hs   = take((size_t)B * 128);
  unsigned* hm   = (unsigned*)take((size_t)B * 128);
  float*    hg   = take((size_t)B * 256);
  float*    xy   = take((size_t)B * 256);
  float*    tb   = take((size_t)B * 128);

  const unsigned NEGINF = 0x007fffffu;  // fenc(-inf)

  auto gemm = [&](const float* X, int lda, const float* W, int K, const float* bias,
                  float* OUT, int M, int Nc, int ldc, int act) {
    const int Kpad = (K + 31) & ~31;
    k_prep_w<<<(unsigned)(((long)Nc * Kpad + 255) / 256), 256, 0, stream>>>(W, Wt, K, Nc, Kpad);
    dim3 g((unsigned)((M + 15) / 16), (unsigned)((Nc + 15) / 16));
    k_gemm<<<g, 32, 0, stream>>>(X, lda, Wt, Kpad, bias, OUT, M, Nc, ldc, act);
  };
  auto fillf = [&](float* p, float v, long n) {
    k_fill_f32<<<(unsigned)((n + 255) / 256), 256, 0, stream>>>(p, v, n);
  };
  auto fillu = [&](unsigned* p, unsigned v, long n) {
    k_fill_u32<<<(unsigned)((n + 255) / 256), 256, 0, stream>>>(p, v, n);
  };
  auto eblocks = [](long n) { return (unsigned)((n + 255) / 256); };

  for (int i = 0; i < 2; ++i) {
    const int*   src = i ? srcB : srcA;
    const int*   dst = i ? dstB : dstA;
    const int*   gid = i ? gidB : gidA;
    const float* nf  = i ? nfB : nfA;
    const float* ef  = i ? efB : efA;
    const float* Wn  = p_Wn  + (size_t)i * 64 * 64;   const float* bn = p_bn + (size_t)i * 64;
    const float* We  = p_We  + (size_t)i * 16 * 16;   const float* be = p_be + (size_t)i * 16;
    const float* Wc  = p_Wc  + (size_t)i * 80 * 80;   const float* bc = p_bc + (size_t)i * 80;
    const float* fc1 = p_fc1 + (size_t)i * 80 * 512;
    const float* al1 = p_al1 + (size_t)i * 512;       const float* ar1 = p_ar1 + (size_t)i * 512;
    const float* gb1 = p_gb1 + (size_t)i * 512;
    const float* Wl1 = p_Wl1 + (size_t)i * 512 * 128; const float* bl1 = p_bl1 + (size_t)i * 128;
    const float* fc2 = p_fc2 + (size_t)i * 128 * 512;
    const float* al2 = p_al2 + (size_t)i * 512;       const float* ar2 = p_ar2 + (size_t)i * 512;
    const float* gb2 = p_gb2 + (size_t)i * 512;
    const float* Wl2 = p_Wl2 + (size_t)i * 512 * 128; const float* bl2 = p_bl2 + (size_t)i * 128;
    const float* wsw = p_wsw + (size_t)i * 128;       const float* wsb = p_wsb + (size_t)i;
    const float* Wp  = p_Wp  + (size_t)i * 256 * 128; const float* bp  = p_bp + (size_t)i * 128;

    // input projections (WMMA)
    gemm(nf, 64, Wn, 64, bn, hn, Nn, 64, 64, 0);
    k_pad_copy<<<eblocks((long)E * 32), 256, 0, stream>>>(ef, efp, E);
    gemm(efp, 32, We, 16, be, he, E, 16, 16, 0);

    // concat(hn[src], he) -> segment_sum over dst (stride-96 padded)
    fillf(xsum, 0.f, (long)Nn * 96);
    k_concat_scatter<<<eblocks((long)E * 80), 256, 0, stream>>>(src, dst, hn, he, xsum, E);

    // EdgeGraphConv linear + relu (WMMA); xc stride 96 with zeroed pad
    fillf(xc, 0.f, (long)Nn * 96);
    gemm(xsum, 96, Wc, 80, bc, xc, Nn, 80, 96, 1);

    auto gat = [&](const float* xin, int lda_in, int din, const float* fc, const float* al_,
                   const float* ar_, const float* gb, const float* Wl, const float* bl,
                   float* xout) {
      gemm(xin, lda_in, fc, din, nullptr, z, Nn, 512, 512, 0);               // z = x@fc (WMMA)
      k_attn_logits<<<(Nn * 4 + 255) / 256, 256, 0, stream>>>(z, al_, ar_, el, er, Nn);
      fillu(mx, NEGINF, (long)Nn * 4);
      k_edge_logit_max<<<eblocks((long)E * 4), 256, 0, stream>>>(src, dst, el, er, ev, mx, E);
      fillf(den, 0.f, (long)Nn * 4);
      k_edge_exp_sum<<<eblocks((long)E * 4), 256, 0, stream>>>(dst, ev, mx, den, E);
      k_edge_norm<<<eblocks((long)E * 4), 256, 0, stream>>>(dst, ev, den, E);
      fillf(rst, 0.f, (long)Nn * 512);
      k_edge_scatter<<<eblocks((long)E * 512), 256, 0, stream>>>(src, dst, ev, z, rst, E);
      k_bias_leaky<<<eblocks((long)Nn * 512), 256, 0, stream>>>(rst, gb, (long)Nn * 512);
      gemm(rst, 512, Wl, 512, bl, xout, Nn, 128, 128, 0);                    // flatten linear (WMMA)
    };
    gat(xc, 96, 80, fc1, al1, ar1, gb1, Wl1, bl1, x1);
    gat(x1, 128, 128, fc2, al2, ar2, gb2, Wl2, bl2, x2);

    // WeightedSumAndMax readout + pool linear
    k_gate<<<(Nn + 255) / 256, 256, 0, stream>>>(x2, wsw, wsb, gate, Nn);
    fillf(hs, 0.f, (long)B * 128);
    fillu(hm, NEGINF, (long)B * 128);
    k_readout<<<eblocks((long)Nn * 128), 256, 0, stream>>>(gid, x2, gate, hs, hm, Nn);
    k_build_hg<<<(B * 256 + 255) / 256, 256, 0, stream>>>(hs, hm, hg, B);
    gemm(hg, 256, Wp, 256, bp, xy + (size_t)i * 128, B, 128, 256, 0);        // branch half of xy
  }

  // final MLP: leaky(xy@Wo1+bo1) @ Wo2 + bo2
  gemm(xy, 256, Wo1, 256, bo1, tb, B, 128, 128, 2);
  k_final<<<1, 64, 0, stream>>>(tb, Wo2, bo2, (float*)d_out, B);
}
